// WindowAttention_22471268893299
// MI455X (gfx1250) — matrix operand order repaired and checked by reading, not compiled, not run on hardware
//
#include <hip/hip_runtime.h>

// ---------------------------------------------------------------------------
// Problem constants
// ---------------------------------------------------------------------------
#define BATCH 32
#define RESOL 56
#define NPIX  3136      // 56*56
#define CDIM  128
#define HEADS 4
#define HD    32
#define NWIN  64        // NQ
#define WA    49
#define NKEY  128       // 2*TOPK
#define MROWS 100352    // BATCH*NPIX
#define OUTROWS 100352  // BATCH*NWIN*WA

typedef __bf16 v16bf __attribute__((ext_vector_type(16)));
typedef __bf16 v2bf  __attribute__((ext_vector_type(2)));
typedef float  v8f   __attribute__((ext_vector_type(8)));

struct U32x8 { uint4 lo, hi; };

// ---------------------------------------------------------------------------
// fp32 -> bf16 conversion helpers (prefer native v_cvt_pk_bf16_f32)
// ---------------------------------------------------------------------------
#if __has_builtin(__builtin_amdgcn_cvt_pk_bf16_f32)
__device__ __forceinline__ unsigned pack2bf(float a, float b) {
  v2bf r = __builtin_amdgcn_cvt_pk_bf16_f32(a, b);
  return __builtin_bit_cast(unsigned, r);
}
__device__ __forceinline__ unsigned short f2bf_u(float f) {
  v2bf r = __builtin_amdgcn_cvt_pk_bf16_f32(f, f);
  return (unsigned short)(__builtin_bit_cast(unsigned, r) & 0xFFFFu);
}
#else
__device__ __forceinline__ unsigned short f2bf_u(float f) {
  return __builtin_bit_cast(unsigned short, static_cast<__bf16>(f));
}
__device__ __forceinline__ unsigned pack2bf(float a, float b) {
  return (unsigned)f2bf_u(a) | ((unsigned)f2bf_u(b) << 16);
}
#endif

__device__ __forceinline__ v8f vzero() {
  v8f z;
#pragma unroll
  for (int e = 0; e < 8; ++e) z[e] = 0.f;
  return z;
}

// A-fragment (16x32 bf16): lane 0-15 = row M, cols {2*c0..2*c0+15} (contiguous)
__device__ __forceinline__ v16bf loadA_bf(const unsigned short* row, int c0) {
  U32x8 s;
  s.lo = *reinterpret_cast<const uint4*>(row + 2 * c0);
  s.hi = *reinterpret_cast<const uint4*>(row + 2 * c0 + 8);
  return __builtin_bit_cast(v16bf, s);
}

// B-fragment (32x16 bf16): lane = col N, 16 contiguous K values
__device__ __forceinline__ v16bf load16bf(const unsigned short* p) {
  U32x8 s;
  s.lo = *reinterpret_cast<const uint4*>(p);
  s.hi = *reinterpret_cast<const uint4*>(p + 8);
  return __builtin_bit_cast(v16bf, s);
}

__device__ __forceinline__ v8f wmma_bf(v16bf a, v16bf b, v8f c) {
  return __builtin_amdgcn_wmma_f32_16x16x32_bf16(false, a, false, b, (short)0, c,
                                                 false, false);
}

// ---------------------------------------------------------------------------
// Kernel 0: bulk fp32 -> bf16 conversion, 8 elements/thread (n must be %8)
// ---------------------------------------------------------------------------
__global__ void cvt_bf16_kernel(const float* __restrict__ in,
                                unsigned short* __restrict__ out, int n8) {
  int t = blockIdx.x * blockDim.x + threadIdx.x;
  if (t >= n8) return;
  const float4* p = reinterpret_cast<const float4*>(in) + (size_t)t * 2;
  float4 a = p[0], b = p[1];
  uint4 r;
  r.x = pack2bf(a.x, a.y);
  r.y = pack2bf(a.z, a.w);
  r.z = pack2bf(b.x, b.y);
  r.w = pack2bf(b.z, b.w);
  reinterpret_cast<uint4*>(out)[t] = r;
}

// ---------------------------------------------------------------------------
// Kernel 1: qkv projection of x0 (100352x128 @ 128x384) with fused routing.
// 4 waves/block, each wave computes a 64x64 tile.  grid = (392, 6)
// ---------------------------------------------------------------------------
__global__ __launch_bounds__(128) void qkv0_kernel(
    const unsigned short* __restrict__ x0_bf,
    const unsigned short* __restrict__ w_bf, const float* __restrict__ qkv_b,
    unsigned short* __restrict__ q_win, unsigned short* __restrict__ k_bf,
    unsigned short* __restrict__ v_bf) {
  const int lane = threadIdx.x & 31;
  const int wave = threadIdx.x >> 5;
  const int m0 = blockIdx.x * 256 + wave * 64;
  const int n0 = blockIdx.y * 64;
  const int lm = lane & 15;
  const int hiHalf = lane >> 4;
  const int c0 = hiHalf ? 8 : 0;
  const int kb = hiHalf ? 16 : 0;

  v8f acc[4][4];
#pragma unroll
  for (int i = 0; i < 4; ++i)
#pragma unroll
    for (int j = 0; j < 4; ++j) acc[i][j] = vzero();

#pragma unroll
  for (int kc = 0; kc < 4; ++kc) {
    v16bf a[4];
#pragma unroll
    for (int mi = 0; mi < 4; ++mi)
      a[mi] = loadA_bf(x0_bf + (size_t)(m0 + mi * 16 + lm) * CDIM + kc * 32, c0);
#pragma unroll
    for (int ni = 0; ni < 4; ++ni) {
      v16bf bf =
          load16bf(w_bf + (size_t)(n0 + ni * 16 + lm) * CDIM + kc * 32 + kb);
#pragma unroll
      for (int mi = 0; mi < 4; ++mi) acc[mi][ni] = wmma_bf(a[mi], bf, acc[mi][ni]);
    }
  }

  const float scale = 0.17677669529663687f;  // HD^-0.5
#pragma unroll
  for (int mi = 0; mi < 4; ++mi) {
#pragma unroll
    for (int r = 0; r < 8; ++r) {
      int m = m0 + mi * 16 + r + hiHalf * 8;
      int b = m / NPIX, pix = m % NPIX;
      int y = pix / RESOL, x = pix % RESOL;
      size_t qbase =
          ((size_t)(b * NWIN + (y / 7) * 8 + (x / 7)) * 64 + (y % 7) * 7 + (x % 7)) *
          CDIM;
      size_t kvbase = (size_t)(b * NPIX + pix) * CDIM;
#pragma unroll
      for (int ni = 0; ni < 4; ++ni) {
        int n = n0 + ni * 16 + lm;
        float val = acc[mi][ni][r] + qkv_b[n];
        if (n < 128)
          q_win[qbase + n] = f2bf_u(val * scale);
        else if (n < 256)
          k_bf[kvbase + (n - 128)] = f2bf_u(val);
        else
          v_bf[kvbase + (n - 256)] = f2bf_u(val);
      }
    }
  }
}

// ---------------------------------------------------------------------------
// Kernel 2: qkv projection of x1 (2048 rows), only k/v columns. grid=(32,4)
// ---------------------------------------------------------------------------
__global__ __launch_bounds__(32) void qkv1_kernel(
    const unsigned short* __restrict__ x1_bf,
    const unsigned short* __restrict__ w_bf, const float* __restrict__ qkv_b,
    unsigned short* __restrict__ k1_bf, unsigned short* __restrict__ v1_bf) {
  const int lane = threadIdx.x & 31;
  const int m0 = blockIdx.x * 64;
  const int n0 = blockIdx.y * 64;  // relative to qkv column 128
  const int lm = lane & 15;
  const int hiHalf = lane >> 4;
  const int c0 = hiHalf ? 8 : 0;
  const int kb = hiHalf ? 16 : 0;

  v8f acc[4][4];
#pragma unroll
  for (int i = 0; i < 4; ++i)
#pragma unroll
    for (int j = 0; j < 4; ++j) acc[i][j] = vzero();

#pragma unroll
  for (int kc = 0; kc < 4; ++kc) {
    v16bf a[4];
#pragma unroll
    for (int mi = 0; mi < 4; ++mi)
      a[mi] = loadA_bf(x1_bf + (size_t)(m0 + mi * 16 + lm) * CDIM + kc * 32, c0);
#pragma unroll
    for (int ni = 0; ni < 4; ++ni) {
      v16bf bf = load16bf(w_bf + (size_t)(128 + n0 + ni * 16 + lm) * CDIM +
                          kc * 32 + kb);
#pragma unroll
      for (int mi = 0; mi < 4; ++mi) acc[mi][ni] = wmma_bf(a[mi], bf, acc[mi][ni]);
    }
  }

#pragma unroll
  for (int mi = 0; mi < 4; ++mi) {
#pragma unroll
    for (int r = 0; r < 8; ++r) {
      int m = m0 + mi * 16 + r + hiHalf * 8;
      int b = m >> 6, p = m & 63;
      size_t base = (size_t)(b * NWIN + p) * CDIM;
#pragma unroll
      for (int ni = 0; ni < 4; ++ni) {
        int nr = n0 + ni * 16 + lm;
        float val = acc[mi][ni][r] + qkv_b[128 + nr];
        if (nr < 128)
          k1_bf[base + nr] = f2bf_u(val);
        else
          v1_bf[base + (nr - 128)] = f2bf_u(val);
      }
    }
  }
}

// ---------------------------------------------------------------------------
// Kernel 3: relative-position-bias MLP (2 -> 32 -> 4), batch independent.
// rpb layout: (NQ, HEADS, WA, 128) fp32
// ---------------------------------------------------------------------------
__global__ void rpb_kernel(const float* __restrict__ coords0,
                           const float* __restrict__ coords1,
                           const float* __restrict__ w1a, const float* __restrict__ b1a,
                           const float* __restrict__ w2a, const float* __restrict__ b2a,
                           const float* __restrict__ w1b, const float* __restrict__ b1b,
                           const float* __restrict__ w2b, const float* __restrict__ b2b,
                           float* __restrict__ rpb) {
  int t = blockIdx.x * blockDim.x + threadIdx.x;
  if (t >= NWIN * WA * NKEY) return;
  int col = t & 127;
  int wa = (t >> 7) % WA;
  int q = t / (128 * WA);
  const float *cw, *w1, *b1, *w2, *b2;
  int tk;
  if (col < 64) { tk = col;      cw = coords0; w1 = w1a; b1 = b1a; w2 = w2a; b2 = b2a; }
  else          { tk = col - 64; cw = coords1; w1 = w1b; b1 = b1b; w2 = w2b; b2 = b2b; }
  const float* cp = cw + (((size_t)q * WA + wa) * 64 + tk) * 2;
  float cx = cp[0], cy = cp[1];
  float out[HEADS];
#pragma unroll
  for (int h = 0; h < HEADS; ++h) out[h] = b2[h];
#pragma unroll 4
  for (int j = 0; j < HD; ++j) {
    float hj = cx * w1[2 * j] + cy * w1[2 * j + 1] + b1[j];
    hj = fmaxf(hj, 0.f);
#pragma unroll
    for (int h = 0; h < HEADS; ++h) out[h] += hj * w2[h * HD + j];
  }
#pragma unroll
  for (int h = 0; h < HEADS; ++h)
    rpb[(((size_t)q * HEADS + h) * WA + wa) * NKEY + col] = out[h];
}

// ---------------------------------------------------------------------------
// Kernel 4: attention.  1 block per (b, window); 4 waves = 4 heads.
// S = Q.K^T (+bias), softmax over 128 keys, O = P.V  — all via bf16 WMMA.
// ---------------------------------------------------------------------------
__global__ __launch_bounds__(128) void attn_kernel(
    const unsigned short* __restrict__ q_win,
    const unsigned short* __restrict__ k_bf,
    const unsigned short* __restrict__ v_bf,
    const unsigned short* __restrict__ k1_bf,
    const unsigned short* __restrict__ v1_bf,
    const int* __restrict__ idx0, const int* __restrict__ idx1,
    const float* __restrict__ rpb, unsigned short* __restrict__ attn_buf) {
  __shared__ __attribute__((aligned(16))) unsigned short Vt[128][136]; // [ch][key]
  __shared__ __attribute__((aligned(16))) unsigned short Pb[HEADS][16][136];

  const int bq = blockIdx.x;
  const int b = bq >> 6, q = bq & 63;
  const int tid = threadIdx.x;
  const int wave = tid >> 5, lane = tid & 31;
  const int lm = lane & 15, hiHalf = lane >> 4;
  const int h = wave;
  const int c0 = hiHalf ? 8 : 0;
  const int kb = hiHalf ? 16 : 0;

  // ---- gather V transposed into LDS (one key per thread, uint4 reads) ----
  {
    const unsigned short* src;
    if (tid < 64) {
      int g = idx0[q * 64 + tid];
      src = v_bf + (size_t)(b * NPIX + g) * CDIM;
    } else {
      int g = idx1[q * 64 + (tid - 64)];
      src = v1_bf + (size_t)(b * NWIN + g) * CDIM;
    }
#pragma unroll
    for (int c8 = 0; c8 < 16; ++c8) {
      uint4 d = reinterpret_cast<const uint4*>(src)[c8];
      const unsigned short* pd = reinterpret_cast<const unsigned short*>(&d);
#pragma unroll
      for (int j = 0; j < 8; ++j) Vt[c8 * 8 + j][tid] = pd[j];
    }
  }
  // ---- per-lane K row pointers (lane = key column of each 16-wide tile) ----
  const unsigned short* krow[8];
#pragma unroll
  for (int nt = 0; nt < 8; ++nt) {
    int j = nt * 16 + lm;
    if (j < 64) {
      int g = idx0[q * 64 + j];
      krow[nt] = k_bf + (size_t)(b * NPIX + g) * CDIM;
    } else {
      int g = idx1[q * 64 + (j - 64)];
      krow[nt] = k1_bf + (size_t)(b * NWIN + g) * CDIM;
    }
  }
  __syncthreads();

  for (int mt = 0; mt < 4; ++mt) {
    // Q A-fragment (rows mt*16..mt*16+15, this head's 32 channels)
    const unsigned short* qrow =
        q_win + ((size_t)bq * 64 + mt * 16 + lm) * CDIM + h * HD;
    v16bf aq = loadA_bf(qrow, c0);

    // ---- S = Q.K^T : 8 key tiles ----
    v8f s[8];
#pragma unroll
    for (int nt = 0; nt < 8; ++nt) {
      v16bf kf = load16bf(krow[nt] + h * HD + kb);
      s[nt] = wmma_bf(aq, kf, vzero());
    }
    // ---- bias ----
#pragma unroll
    for (int nt = 0; nt < 8; ++nt)
#pragma unroll
      for (int r = 0; r < 8; ++r) {
        int m = mt * 16 + r + hiHalf * 8;
        float bias = (m < WA)
            ? rpb[(((size_t)q * HEADS + h) * WA + m) * NKEY + nt * 16 + lm]
            : 0.f;
        s[nt][r] += bias;
      }
    // ---- softmax over 128 keys (row m lives in 16 lanes x 8 frags) ----
#pragma unroll
    for (int r = 0; r < 8; ++r) {
      float mx = -1e30f;
#pragma unroll
      for (int nt = 0; nt < 8; ++nt) mx = fmaxf(mx, s[nt][r]);
#pragma unroll
      for (int d = 1; d < 16; d <<= 1) mx = fmaxf(mx, __shfl_xor(mx, d, 32));
      float sum = 0.f;
#pragma unroll
      for (int nt = 0; nt < 8; ++nt) {
        float e = __expf(s[nt][r] - mx);
        s[nt][r] = e;
        sum += e;
      }
#pragma unroll
      for (int d = 1; d < 16; d <<= 1) sum += __shfl_xor(sum, d, 32);
      float inv = 1.f / sum;
#pragma unroll
      for (int nt = 0; nt < 8; ++nt) s[nt][r] *= inv;
    }
    // ---- C-layout -> A-layout transpose of P through wave-private LDS ----
#pragma unroll
    for (int nt = 0; nt < 8; ++nt)
#pragma unroll
      for (int r = 0; r < 8; ++r)
        Pb[h][r + hiHalf * 8][nt * 16 + lm] = f2bf_u(s[nt][r]);

    // ---- O = P.V ----
    v8f o[2] = {vzero(), vzero()};
#pragma unroll
    for (int kc = 0; kc < 4; ++kc) {
      v16bf ap = loadA_bf(&Pb[h][lm][kc * 32], c0);
#pragma unroll
      for (int n2 = 0; n2 < 2; ++n2) {
        v16bf vf = load16bf(&Vt[h * HD + n2 * 16 + lm][kc * 32 + kb]);
        o[n2] = wmma_bf(ap, vf, o[n2]);
      }
    }
    // ---- write valid rows ----
#pragma unroll
    for (int n2 = 0; n2 < 2; ++n2)
#pragma unroll
      for (int r = 0; r < 8; ++r) {
        int m = mt * 16 + r + hiHalf * 8;
        if (m < WA)
          attn_buf[((size_t)bq * WA + m) * CDIM + h * HD + n2 * 16 + lm] =
              f2bf_u(o[n2][r]);
      }
  }
}

// ---------------------------------------------------------------------------
// Kernel 5: output projection (100352x128 @ 128x128) + bias -> fp32 out.
// grid = (392, 2), block 128 (4 waves x 64x64 tiles)
// ---------------------------------------------------------------------------
__global__ __launch_bounds__(128) void proj_kernel(
    const unsigned short* __restrict__ attn_buf,
    const unsigned short* __restrict__ w_bf, const float* __restrict__ proj_b,
    float* __restrict__ out) {
  const int lane = threadIdx.x & 31;
  const int wave = threadIdx.x >> 5;
  const int m0 = blockIdx.x * 256 + wave * 64;
  const int n0 = blockIdx.y * 64;
  const int lm = lane & 15;
  const int hiHalf = lane >> 4;
  const int c0 = hiHalf ? 8 : 0;
  const int kb = hiHalf ? 16 : 0;

  v8f acc[4][4];
#pragma unroll
  for (int i = 0; i < 4; ++i)
#pragma unroll
    for (int j = 0; j < 4; ++j) acc[i][j] = vzero();

#pragma unroll
  for (int kc = 0; kc < 4; ++kc) {
    v16bf a[4];
#pragma unroll
    for (int mi = 0; mi < 4; ++mi)
      a[mi] = loadA_bf(attn_buf + (size_t)(m0 + mi * 16 + lm) * CDIM + kc * 32, c0);
#pragma unroll
    for (int ni = 0; ni < 4; ++ni) {
      v16bf bf =
          load16bf(w_bf + (size_t)(n0 + ni * 16 + lm) * CDIM + kc * 32 + kb);
#pragma unroll
      for (int mi = 0; mi < 4; ++mi) acc[mi][ni] = wmma_bf(a[mi], bf, acc[mi][ni]);
    }
  }

#pragma unroll
  for (int mi = 0; mi < 4; ++mi)
#pragma unroll
    for (int ni = 0; ni < 4; ++ni)
#pragma unroll
      for (int r = 0; r < 8; ++r) {
        int m = m0 + mi * 16 + r + hiHalf * 8;
        int n = n0 + ni * 16 + lm;
        out[(size_t)m * CDIM + n] = acc[mi][ni][r] + proj_b[n];
      }
}

// ---------------------------------------------------------------------------
// Host launcher
// ---------------------------------------------------------------------------
extern "C" void kernel_launch(void* const* d_in, const int* in_sizes, int n_in,
                              void* d_out, int out_size, void* d_ws,
                              size_t ws_size, hipStream_t stream) {
  const float* x0      = (const float*)d_in[0];
  const float* x1      = (const float*)d_in[1];
  const float* qkv_w   = (const float*)d_in[2];
  const float* qkv_b   = (const float*)d_in[3];
  const float* proj_w  = (const float*)d_in[4];
  const float* proj_b  = (const float*)d_in[5];
  const float* rpb0_w1 = (const float*)d_in[6];
  const float* rpb0_b1 = (const float*)d_in[7];
  const float* rpb0_w2 = (const float*)d_in[8];
  const float* rpb0_b2 = (const float*)d_in[9];
  const float* rpb1_w1 = (const float*)d_in[10];
  const float* rpb1_b1 = (const float*)d_in[11];
  const float* rpb1_w2 = (const float*)d_in[12];
  const float* rpb1_b2 = (const float*)d_in[13];
  const float* coords0 = (const float*)d_in[14];
  const float* coords1 = (const float*)d_in[15];
  const int*   idx0    = (const int*)d_in[16];
  const int*   idx1    = (const int*)d_in[17];

  char* ws = (char*)d_ws;
  size_t off = 0;
  auto alloc = [&](size_t bytes) -> void* {
    void* p = ws + off;
    off += (bytes + 255) & ~(size_t)255;
    return p;
  };
  const size_t qwin_bytes = (size_t)BATCH * NWIN * 64 * CDIM * 2;
  unsigned short* q_win    = (unsigned short*)alloc(qwin_bytes);
  unsigned short* k_bf     = (unsigned short*)alloc((size_t)BATCH * NPIX * CDIM * 2);
  unsigned short* v_bf     = (unsigned short*)alloc((size_t)BATCH * NPIX * CDIM * 2);
  unsigned short* k1_bf    = (unsigned short*)alloc((size_t)BATCH * NWIN * CDIM * 2);
  unsigned short* v1_bf    = (unsigned short*)alloc((size_t)BATCH * NWIN * CDIM * 2);
  float*          rpb      = (float*)alloc((size_t)NWIN * HEADS * WA * NKEY * 4);
  unsigned short* attn_buf = (unsigned short*)alloc((size_t)OUTROWS * CDIM * 2);
  unsigned short* x0_bf    = (unsigned short*)alloc((size_t)MROWS * CDIM * 2);
  unsigned short* x1_bf    = (unsigned short*)alloc((size_t)BATCH * NWIN * CDIM * 2);
  unsigned short* wq_bf    = (unsigned short*)alloc((size_t)384 * CDIM * 2);
  unsigned short* wp_bf    = (unsigned short*)alloc((size_t)CDIM * CDIM * 2);

  // zero the padded query-window rows (rows 49..63 of every window)
  hipMemsetAsync(q_win, 0, qwin_bytes, stream);

  // pre-convert fp32 operands to bf16 (all sizes divisible by 8)
  const int nx0 = MROWS * CDIM / 8;
  cvt_bf16_kernel<<<(nx0 + 255) / 256, 256, 0, stream>>>(x0, x0_bf, nx0);
  const int nx1 = BATCH * NWIN * CDIM / 8;
  cvt_bf16_kernel<<<(nx1 + 255) / 256, 256, 0, stream>>>(x1, x1_bf, nx1);
  const int nwq = 384 * CDIM / 8;
  cvt_bf16_kernel<<<(nwq + 255) / 256, 256, 0, stream>>>(qkv_w, wq_bf, nwq);
  const int nwp = CDIM * CDIM / 8;
  cvt_bf16_kernel<<<(nwp + 255) / 256, 256, 0, stream>>>(proj_w, wp_bf, nwp);

  qkv0_kernel<<<dim3(MROWS / 256, 384 / 64), 128, 0, stream>>>(
      x0_bf, wq_bf, qkv_b, q_win, k_bf, v_bf);
  qkv1_kernel<<<dim3((BATCH * NWIN) / 64, 256 / 64), 32, 0, stream>>>(
      x1_bf, wq_bf, qkv_b, k1_bf, v1_bf);
  rpb_kernel<<<(NWIN * WA * NKEY + 255) / 256, 256, 0, stream>>>(
      coords0, coords1, rpb0_w1, rpb0_b1, rpb0_w2, rpb0_b2, rpb1_w1, rpb1_b1,
      rpb1_w2, rpb1_b2, rpb);
  attn_kernel<<<BATCH * NWIN, 128, 0, stream>>>(
      q_win, k_bf, v_bf, k1_bf, v1_bf, idx0, idx1, rpb, attn_buf);
  proj_kernel<<<dim3(MROWS / 256, 128 / 64), 128, 0, stream>>>(
      attn_buf, wp_bf, proj_b, (float*)d_out);
}
